// GATBlock_85779086836240
// MI455X (gfx1250) — compile-verified, matrix operand order checked by reading, not run on hardware
//
#include <hip/hip_runtime.h>

// ---------------- problem constants (fixed by the reference) ----------------
constexpr int N_NODES = 50000;
constexpr int E_EDGES = 400000;
constexpr int ETOT    = E_EDGES + N_NODES;   // 450000 (with self loops)
constexpr int IN_F    = 128;
constexpr int H_HEADS = 16;
constexpr int C_CH    = 32;
constexpr int HC      = H_HEADS * C_CH;      // 512
constexpr int WCOLS   = HC + HC + C_CH;      // 1056 (W_l | W_r | W_res fused)
constexpr float NEG_SLOPE = 0.2f;

constexpr int TILES_M = N_NODES / 16;        // 3125
constexpr int TILES_N = WCOLS / 16;          // 66
constexpr int NB      = 6;                   // column tiles per wave (96 cols)
constexpr int NGRP    = TILES_N / NB;        // 11 (exact)
constexpr int TOT_TASKS = TILES_M * NGRP;    // 34375 wave-tasks

// ---------------- CDNA5 WMMA types ----------------
typedef __attribute__((ext_vector_type(16))) __bf16 v16bf;
typedef __attribute__((ext_vector_type(8)))  float  v8f;

__device__ __forceinline__ unsigned short f32_to_bf16(float f) {
  unsigned int u = __float_as_uint(f);
  unsigned int r = u + 0x7FFFu + ((u >> 16) & 1u);   // round-to-nearest-even
  return (unsigned short)(r >> 16);
}

union FragU { v16bf v; uint4 q[2]; };

// CDNA5 16-bit A/B fragment layout: lanes 0-15 hold K in {0..7, 16..23},
// lanes 16-31 hold K in {8..15, 24..31}. Each 8-elem run = one 16B load.
__device__ __forceinline__ v16bf load_frag(const unsigned short* rowk, int kbase) {
  FragU f;
  f.q[0] = *(const uint4*)(rowk + kbase);
  f.q[1] = *(const uint4*)(rowk + kbase + 16);
  return f.v;
}

__device__ __forceinline__ void atomicMaxFloat(float* addr, float v) {
  if (v >= 0.0f) atomicMax((int*)addr, __float_as_int(v));
  else           atomicMin((unsigned int*)addr, (unsigned int)__float_as_int(v));
}

// ---------------- kernel 1: fp32 -> bf16 convert + weight transpose ----------------
__global__ void k_convert(const float* __restrict__ x,
                          const float* __restrict__ Wl,
                          const float* __restrict__ Wr,
                          const float* __restrict__ Wres,
                          unsigned short* __restrict__ Xb,
                          unsigned short* __restrict__ Wt) {
  int i = blockIdx.x * blockDim.x + threadIdx.x;
  if (i < N_NODES * IN_F) Xb[i] = f32_to_bf16(x[i]);
  if (i < WCOLS * IN_F) {
    int n = i / IN_F, k = i % IN_F;
    float v;
    if (n < HC)            v = Wl[k * HC + n];
    else if (n < 2 * HC)   v = Wr[k * HC + (n - HC)];
    else                   v = Wres[k * C_CH + (n - 2 * HC)];
    Wt[i] = f32_to_bf16(v);
  }
}

// ---------------- kernel 2: init segment max / sum ----------------
__global__ void k_init_ms(float* __restrict__ m, float* __restrict__ s) {
  int i = blockIdx.x * blockDim.x + threadIdx.x;
  if (i < N_NODES * H_HEADS) { m[i] = -1e30f; s[i] = 0.0f; }
}

// ---------------- kernel 3: fused WMMA GEMM  [50000x128] x [128x1056] ----------------
// One wave: 16-row tile x 6 column tiles. A fragments (4 x v16bf) stay in
// registers and are reused for all 6 column tiles -> 24 WMMAs per wave.
// Destination buffer split (xl|xr|xres) is wave-uniform (gc0 uniform).
__global__ void k_gemm_wmma(const unsigned short* __restrict__ Xb,
                            const unsigned short* __restrict__ Wt,
                            float* __restrict__ xl,
                            float* __restrict__ xr,
                            float* __restrict__ xres) {
  int wave = threadIdx.x >> 5;
  int lane = threadIdx.x & 31;
  int task = blockIdx.x * (blockDim.x >> 5) + wave;
  if (task >= TOT_TASKS) return;   // uniform per wave: EXEC stays all-ones

  int tm  = task / NGRP;
  int grp = task % NGRP;
  int row0 = tm * 16;
  int kbase = (lane < 16) ? 0 : 8;
  int col   = lane & 15;
  int mbase = (lane < 16) ? 0 : 8;

  const unsigned short* Aptr = Xb + (size_t)(row0 + col) * IN_F;
  v16bf a[4];
#pragma unroll
  for (int k4 = 0; k4 < 4; ++k4) a[k4] = load_frag(Aptr + k4 * 32, kbase);

#pragma unroll
  for (int j = 0; j < NB; ++j) {
    int gc0 = grp * (NB * 16) + j * 16;           // wave-uniform base column
    const unsigned short* Bptr = Wt + (size_t)(gc0 + col) * IN_F;

    v8f acc = {};
#pragma unroll
    for (int k4 = 0; k4 < 4; ++k4) {
      v16bf b = load_frag(Bptr + k4 * 32, kbase);
      acc = __builtin_amdgcn_wmma_f32_16x16x32_bf16(
          /*neg_a=*/false, a[k4], /*neg_b=*/false, b,
          /*c_mod=*/(short)0, acc, /*reuse_a=*/false, /*reuse_b=*/false);
    }

    // wave-uniform destination select (scalar branch, no exec divergence)
    float* base; int ld; int c0;
    if (gc0 < HC)           { base = xl;   ld = HC;   c0 = gc0; }
    else if (gc0 < 2 * HC)  { base = xr;   ld = HC;   c0 = gc0 - HC; }
    else                    { base = xres; ld = C_CH; c0 = gc0 - 2 * HC; }

#pragma unroll
    for (int r = 0; r < 8; ++r)
      base[(size_t)(row0 + mbase + r) * ld + c0 + col] = acc[r];
  }
}

// ---------------- kernel 4: GATv2 scores + segment max ----------------
__global__ void k_edge_scores(const int* __restrict__ ei,
                              const float* __restrict__ xl,
                              const float* __restrict__ xr,
                              const float* __restrict__ att,
                              float* __restrict__ alpha,
                              float* __restrict__ m) {
  int gid = blockIdx.x * blockDim.x + threadIdx.x;
  if (gid >= ETOT * H_HEADS) return;
  int e = gid >> 4, h = gid & 15;
  int src, dst;
  if (e < E_EDGES) { src = ei[e]; dst = ei[E_EDGES + e]; }
  else             { src = dst = e - E_EDGES; }
  const float* pl = xl + (size_t)src * HC + h * C_CH;
  const float* pr = xr + (size_t)dst * HC + h * C_CH;
  const float* pa = att + h * C_CH;
  float sum = 0.0f;
#pragma unroll
  for (int c = 0; c < C_CH; c += 4) {
    float4 a4 = *(const float4*)(pl + c);
    float4 b4 = *(const float4*)(pr + c);
    float4 w4 = *(const float4*)(pa + c);
    float v;
    v = a4.x + b4.x; sum += (v > 0.f ? v : NEG_SLOPE * v) * w4.x;
    v = a4.y + b4.y; sum += (v > 0.f ? v : NEG_SLOPE * v) * w4.y;
    v = a4.z + b4.z; sum += (v > 0.f ? v : NEG_SLOPE * v) * w4.z;
    v = a4.w + b4.w; sum += (v > 0.f ? v : NEG_SLOPE * v) * w4.w;
  }
  alpha[gid] = sum;
  atomicMaxFloat(&m[dst * H_HEADS + h], sum);
}

// ---------------- kernel 5: zero the aggregation buffer (aliases xr) ----------------
__global__ void k_zero(float* __restrict__ p, int n) {
  int i = blockIdx.x * blockDim.x + threadIdx.x;
  if (i < n) p[i] = 0.0f;
}

// ---------------- kernel 6: exp(e - max) + segment sum ----------------
__global__ void k_edge_exp(const int* __restrict__ ei,
                           float* __restrict__ alpha,
                           const float* __restrict__ m,
                           float* __restrict__ s) {
  int gid = blockIdx.x * blockDim.x + threadIdx.x;
  if (gid >= ETOT * H_HEADS) return;
  int e = gid >> 4, h = gid & 15;
  int dst = (e < E_EDGES) ? ei[E_EDGES + e] : (e - E_EDGES);
  float ex = __expf(alpha[gid] - m[dst * H_HEADS + h]);
  alpha[gid] = ex;
  atomicAdd(&s[dst * H_HEADS + h], ex);
}

// ---------------- kernel 7: normalize alpha + weighted scatter aggregate ----------------
__global__ void k_edge_aggregate(const int* __restrict__ ei,
                                 const float* __restrict__ xl,
                                 const float* __restrict__ s,
                                 float* __restrict__ alpha,
                                 float* __restrict__ accum) {
  int gid = blockIdx.x * blockDim.x + threadIdx.x;
  if (gid >= ETOT * H_HEADS) return;
  int e = gid >> 4, h = gid & 15;
  int src, dst;
  if (e < E_EDGES) { src = ei[e]; dst = ei[E_EDGES + e]; }
  else             { src = dst = e - E_EDGES; }
  float a = alpha[gid] / (s[dst * H_HEADS + h] + 1e-16f);
  alpha[gid] = a;                         // final alpha output
  const float* pl = xl + (size_t)src * HC + h * C_CH;
  float* pd = accum + (size_t)dst * HC + h * C_CH;
#pragma unroll
  for (int c = 0; c < C_CH; ++c)
    atomicAdd(&pd[c], a * pl[c]);         // accum (102 MB) lives in 192 MB L2
}

// ---------------- kernel 8: head mean + bias + ReLU + residual ----------------
__global__ void k_finalize(const float* __restrict__ accum,
                           const float* __restrict__ xres,
                           const float* __restrict__ bias,
                           float* __restrict__ xout) {
  int i = blockIdx.x * blockDim.x + threadIdx.x;
  if (i >= N_NODES * C_CH) return;
  int n = i >> 5, c = i & 31;
  const float* p = accum + (size_t)n * HC + c;
  float sum = 0.0f;
#pragma unroll
  for (int h = 0; h < H_HEADS; ++h) sum += p[h * C_CH];
  float g = sum * (1.0f / H_HEADS) + bias[c];
  g = g > 0.0f ? g : 0.0f;
  xout[i] = g + xres[i];
}

// ---------------- host launcher ----------------
extern "C" void kernel_launch(void* const* d_in, const int* in_sizes, int n_in,
                              void* d_out, int out_size, void* d_ws, size_t ws_size,
                              hipStream_t stream) {
  const float* x    = (const float*)d_in[0];
  const int*   ei   = (const int*)d_in[1];
  // d_in[2] edge_attr and d_in[3] batch are unused by the reference
  const float* Wl   = (const float*)d_in[4];
  const float* Wr   = (const float*)d_in[5];
  const float* att  = (const float*)d_in[6];
  const float* bias = (const float*)d_in[7];
  const float* Wres = (const float*)d_in[8];

  char* ws = (char*)d_ws;
  size_t off = 0;
  auto carve = [&](size_t bytes) -> void* {
    void* p = ws + off;
    off = (off + bytes + 255) & ~(size_t)255;
    return p;
  };
  unsigned short* Xb = (unsigned short*)carve((size_t)N_NODES * IN_F * 2);
  unsigned short* Wt = (unsigned short*)carve((size_t)WCOLS * IN_F * 2);
  float* xl   = (float*)carve((size_t)N_NODES * HC * 4);
  float* xr   = (float*)carve((size_t)N_NODES * HC * 4);   // reused as accum
  float* xres = (float*)carve((size_t)N_NODES * C_CH * 4);
  float* mbuf = (float*)carve((size_t)N_NODES * H_HEADS * 4);
  float* sbuf = (float*)carve((size_t)N_NODES * H_HEADS * 4);
  float* accum = xr;   // alias: xr is dead after k_edge_scores

  float* xout  = (float*)d_out;                              // [N, 32]
  float* alpha = xout + (size_t)N_NODES * C_CH;              // [Etot, 16]

  const int B = 256;
  int g_conv = (N_NODES * IN_F + B - 1) / B;                 // 25000
  int g_ms   = (N_NODES * H_HEADS + B - 1) / B;              // 3125
  int g_gemm = (TOT_TASKS + 8 - 1) / 8;                      // 4297 (8 waves/block)
  int g_edge = (ETOT * H_HEADS + B - 1) / B;                 // 28125
  int g_zero = (N_NODES * HC + B - 1) / B;                   // 100000
  int g_fin  = (N_NODES * C_CH + B - 1) / B;                 // 6250

  k_convert<<<g_conv, B, 0, stream>>>(x, Wl, Wr, Wres, Xb, Wt);
  k_init_ms<<<g_ms, B, 0, stream>>>(mbuf, sbuf);
  k_gemm_wmma<<<g_gemm, B, 0, stream>>>(Xb, Wt, xl, xr, xres);
  k_edge_scores<<<g_edge, B, 0, stream>>>(ei, xl, xr, att, alpha, mbuf);
  k_zero<<<g_zero, B, 0, stream>>>(accum, N_NODES * HC);
  k_edge_exp<<<g_edge, B, 0, stream>>>(ei, alpha, mbuf, sbuf);
  k_edge_aggregate<<<g_edge, B, 0, stream>>>(ei, xl, sbuf, alpha, accum);
  k_finalize<<<g_fin, B, 0, stream>>>(accum, xres, bias, xout);
}
